// RecurrentBlock_48103633715278
// MI455X (gfx1250) — matrix operand order repaired and checked by reading, not compiled, not run on hardware
//
#include <hip/hip_runtime.h>
#include <math.h>

typedef __attribute__((ext_vector_type(16))) __bf16 v16bf;
typedef __attribute__((ext_vector_type(8)))  __bf16 v8bf;
typedef __attribute__((ext_vector_type(8)))  float  v8f;

#define B_     4
#define L_     2048
#define WIDTH_ 2560
#define LRU_   2560
#define NB_    10
#define BD_    256
#define TW_    4
#define M_     (B_ * L_)            // 8192 rows for all GEMMs

#define WMMA_BF16(a, b, c) \
  __builtin_amdgcn_wmma_f32_16x16x32_bf16(false, (a), false, (b), (short)0, (c), false, false)

// ---------------------------------------------------------------------------
// Fragment load helper (contiguous k order used by this code for A and B, wave32)
//  A (16x32 bf16): lane<16 -> M=lane,   K=0..15
//                  lane>=16 -> M=lane-16, K=16..31
//  B (32x16 bf16): lane<16 -> N=lane,   K=0..15 ; lane>=16 -> N=lane-16, K=16..31
// ---------------------------------------------------------------------------
__device__ __forceinline__ v16bf load_afrag(const __bf16* __restrict__ p) {
  v8bf a0 = *(const v8bf*)p;          // K {0..7} or {16..23}
  v8bf a1 = *(const v8bf*)(p + 8);    // K {8..15} or {24..31}
  v16bf r;
#pragma unroll
  for (int i = 0; i < 8; ++i) { r[i] = a0[i]; r[i + 8] = a1[i]; }
  return r;
}

// ---------------------------------------------------------------------------
// Conversion kernels
// ---------------------------------------------------------------------------
__global__ void k_cvt_bf16(const float* __restrict__ in, __bf16* __restrict__ out, int n) {
  int i = blockIdx.x * 256 + threadIdx.x;
  if (i < n) out[i] = (__bf16)in[i];
}

// in: K x N (row-major f32) -> out: N x K (row-major bf16) == B transposed
__global__ void k_cvt_t(const float* __restrict__ in, __bf16* __restrict__ out, int K, int N) {
  int i = blockIdx.x * 256 + threadIdx.x;
  if (i >= K * N) return;
  int k = i / N, n = i % N;                     // coalesced read
  out[(size_t)n * K + k] = (__bf16)in[i];
}

// (NB, bd, bd) with w[h][i][j] -> wt[h][j][i] bf16
__global__ void k_cvt_blk_t(const float* __restrict__ in, __bf16* __restrict__ out) {
  int i = blockIdx.x * 256 + threadIdx.x;
  if (i >= NB_ * BD_ * BD_) return;
  int h = i / (BD_ * BD_);
  int r = (i / BD_) % BD_;
  int c = i % BD_;
  out[(size_t)h * BD_ * BD_ + (size_t)c * BD_ + r] = (__bf16)in[i];
}

__device__ __forceinline__ float gelu_tanh(float x) {
  float u = 0.7978845608028654f * (x + 0.044715f * x * x * x);
  return 0.5f * x * (1.0f + tanhf(u));
}
__device__ __forceinline__ float sigmoidf_(float x) {
  return 1.0f / (1.0f + expf(-x));
}

// ---------------------------------------------------------------------------
// Generalized LDS-staged, double-buffered WMMA GEMM (all three matrix stages).
//   C1 = epi(A @ B1 + bias1)   [+ C2 when DUAL]
// Strided (lda/ldb/ldc) + batched over blockIdx.z (block-diagonal gates).
// Block = 128 threads (4 waves). Wave w computes rows [blockIdx.x*64 + w*16,+16),
// cols [blockIdx.y*64,+64) of batch z. Per K-step the block cooperatively stages
// the 64x32 B tile(s) into LDS (one 32B chunk per thread) -> each B tile is read
// from L2 once per BLOCK (4x cut) and WMMA operands come from ds_load. Staging
// registers are disjoint from WMMA operands so global latency overlaps the
// whole WMMA group; one __syncthreads per K-step; 16KB LDS of 320KB/WGP.
// EPI: 0 = bias; 1 = gelu(C1)+bias(C2); 2 = sigmoid(C1) & sigmoid(C2)
// ---------------------------------------------------------------------------
template <int DUAL, int EPI>
__global__ void k_gemm_lds(const __bf16* __restrict__ A0, int lda,
                           const __bf16* __restrict__ B10, const __bf16* __restrict__ B20, int ldb,
                           const float* __restrict__ bias10, const float* __restrict__ bias20,
                           float* __restrict__ C10, float* __restrict__ C20, int ldc,
                           int K, int zAoff, int zBoff, int zCoff, int zBias) {
  __shared__ __align__(32) __bf16 sB1[2 * 64 * 32];
  __shared__ __align__(32) __bf16 sB2[DUAL ? 2 * 64 * 32 : 32];

  const int z = blockIdx.z;
  const __bf16* A     = A0 + (size_t)z * zAoff;
  const __bf16* B1t   = B10 + (size_t)z * zBoff;
  const __bf16* B2t   = DUAL ? (B20 + (size_t)z * zBoff) : (const __bf16*)0;
  const float*  bias1 = bias10 + (size_t)z * zBias;
  const float*  bias2 = DUAL ? (bias20 + (size_t)z * zBias) : (const float*)0;
  float* C1 = C10 + (size_t)z * zCoff;
  float* C2 = DUAL ? (C20 + (size_t)z * zCoff) : (float*)0;

  const int tid  = threadIdx.x;
  const int wave = tid >> 5;
  const int lane = tid & 31;
  const int n    = lane & 15;
  const int half = lane >> 4;
  const int row0 = blockIdx.x * 64 + wave * 16;
  const int col0 = blockIdx.y * 64;

  // cooperative staging map: thread -> (tile col sr, k-chunk sk), 32B each
  const int sr = tid >> 1;             // 0..63
  const int sk = (tid & 1) * 16;       // 0 or 16
  const __bf16* g1 = B1t + (size_t)(col0 + sr) * ldb + sk;
  const __bf16* g2 = DUAL ? (B2t + (size_t)(col0 + sr) * ldb + sk) : (const __bf16*)0;

  v8f acc1[4] = {}, acc2[4] = {};
  const __bf16* Ap = A + (size_t)(row0 + n) * lda + half * 16;
  const int lbase = n * 32 + half * 16;          // LDS frag base (elements)

  // prologue: stage K-step 0
  v16bf r1 = *(const v16bf*)g1;
  v16bf r2;
  if (DUAL) r2 = *(const v16bf*)g2;
  *(v16bf*)&sB1[sr * 32 + sk] = r1;
  if (DUAL) *(v16bf*)&sB2[sr * 32 + sk] = r2;
  v16bf af = load_afrag(Ap);
  __syncthreads();

  int buf = 0;
  for (int k = 0; k < K; k += 32) {
    const int kn    = k + 32;
    const bool more = kn < K;
    // issue next-stage global loads into registers (in flight during compute)
    if (more) {
      r1 = *(const v16bf*)(g1 + kn);
      if (DUAL) r2 = *(const v16bf*)(g2 + kn);
    }
    v16bf afn = load_afrag(Ap + (more ? kn : k));

    // compute current stage from LDS
    const int cb = buf * (64 * 32);
#pragma unroll
    for (int t = 0; t < 4; ++t) {
      v16bf b1 = *(const v16bf*)&sB1[cb + t * 16 * 32 + lbase];
      acc1[t] = WMMA_BF16(af, b1, acc1[t]);
      if (DUAL) {
        v16bf b2 = *(const v16bf*)&sB2[cb + t * 16 * 32 + lbase];
        acc2[t] = WMMA_BF16(af, b2, acc2[t]);
      }
    }
    af = afn;

    if (more) {
      const int nb = (buf ^ 1) * (64 * 32);
      *(v16bf*)&sB1[nb + sr * 32 + sk] = r1;
      if (DUAL) *(v16bf*)&sB2[nb + sr * 32 + sk] = r2;
      buf ^= 1;
    }
    __syncthreads();   // next-stage store complete before anyone reads it
  }

#pragma unroll
  for (int t = 0; t < 4; ++t) {
    int col = col0 + t * 16 + n;
    float b1v = bias1[col];
    float b2v = DUAL ? bias2[col] : 0.0f;
#pragma unroll
    for (int r = 0; r < 8; ++r) {
      size_t o = (size_t)(row0 + r + half * 8) * ldc + col;
      float v1 = acc1[t][r] + b1v;
      if (EPI == 1)      C1[o] = gelu_tanh(v1);
      else if (EPI == 2) C1[o] = sigmoidf_(v1);
      else               C1[o] = v1;
      if (DUAL) {
        float v2 = acc2[t][r] + b2v;
        C2[o] = (EPI == 2) ? sigmoidf_(v2) : v2;
      }
    }
  }
}

// ---------------------------------------------------------------------------
// Causal width-4 depthwise conv with segment-boundary masking.
// ---------------------------------------------------------------------------
__global__ void k_conv(const float* __restrict__ hpre, const float* __restrict__ cw,
                       const float* __restrict__ cb, const int* __restrict__ segp,
                       float* __restrict__ hconv, __bf16* __restrict__ hconv_bf) {
  size_t i = (size_t)blockIdx.x * 256 + threadIdx.x;
  if (i >= (size_t)M_ * LRU_) return;
  int c = (int)(i % LRU_);
  size_t bt = i / LRU_;
  int t = (int)(bt % L_);

  float acc = cb[c] + cw[(TW_ - 1) * LRU_ + c] * hpre[i];
  float mask = 1.0f;
#pragma unroll
  for (int s = 1; s < TW_; ++s) {
    if (t - s < 0) break;
    mask *= (segp[t - s + 1] != 0) ? 1.0f : 0.0f;
    acc += mask * cw[(TW_ - 1 - s) * LRU_ + c] * hpre[i - (size_t)s * LRU_];
  }
  hconv[i] = acc;
  hconv_bf[i] = (__bf16)acc;
}

// ---------------------------------------------------------------------------
// RG-LRU sequential scan, fused with (h * y) and bf16 convert.
// Software-pipelined: loads for t+1 issue before the transcendental chain of t;
// prefetch runs 8 timesteps ahead so each step is VALU-bound, not L2-latency.
// ---------------------------------------------------------------------------
__global__ void k_scan(const float* __restrict__ Gx, const float* __restrict__ Ga,
                       const float* __restrict__ hconv, const float* __restrict__ y,
                       const float* __restrict__ a_param, const int* __restrict__ segp,
                       __bf16* __restrict__ hy_bf) {
  int idx = blockIdx.x * 256 + threadIdx.x;
  if (idx >= B_ * LRU_) return;
  int b = idx / LRU_, c = idx % LRU_;
  float ap = a_param[c];
  float sp8 = 8.0f * ((ap > 20.0f) ? ap : log1pf(expf(ap)));   // 8*softplus(a_param)
  float h = 0.0f;
  size_t base = (size_t)b * L_ * LRU_ + c;

  float ga = Ga[base], gx = Gx[base], xv = hconv[base], yv = y[base];
  int sp = segp[0];
  for (int t = 0; t < L_; ++t) {
    size_t o = base + (size_t)t * LRU_;
    // issue next-step loads early
    size_t onext = o + ((t + 1 < L_) ? (size_t)LRU_ : 0);
    float gan = Ga[onext], gxn = Gx[onext], xvn = hconv[onext], yvn = y[onext];
    int spn = segp[(t + 1 < L_) ? t + 1 : t];
    if (t + 8 < L_) {
      size_t op = o + (size_t)8 * LRU_;
      __builtin_prefetch(Ga + op, 0, 3);
      __builtin_prefetch(Gx + op, 0, 3);
      __builtin_prefetch(hconv + op, 0, 3);
      __builtin_prefetch(y + op, 0, 3);
    }
    float la = -sp8 * ga;
    bool reset = (sp == 0);
    float a    = reset ? 0.0f : expf(la);
    float mult = reset ? 1.0f : sqrtf(fmaxf(1.0f - expf(2.0f * la), 0.0f));
    float xn = xv * gx * mult;
    h = fmaf(a, h, xn);
    hy_bf[o] = (__bf16)(h * yv);
    ga = gan; gx = gxn; xv = xvn; yv = yvn; sp = spn;
  }
}

// ---------------------------------------------------------------------------
extern "C" void kernel_launch(void* const* d_in, const int* in_sizes, int n_in,
                              void* d_out, int out_size, void* d_ws, size_t ws_size,
                              hipStream_t stream) {
  const float* x       = (const float*)d_in[0];
  const int*   segp    = (const int*)  d_in[1];
  const float* w_y     = (const float*)d_in[2];
  const float* b_y     = (const float*)d_in[3];
  const float* w_x     = (const float*)d_in[4];
  const float* b_x     = (const float*)d_in[5];
  const float* conv_w  = (const float*)d_in[6];
  const float* conv_b  = (const float*)d_in[7];
  const float* wi_gate = (const float*)d_in[8];
  const float* bi_gate = (const float*)d_in[9];
  const float* wa_gate = (const float*)d_in[10];
  const float* ba_gate = (const float*)d_in[11];
  const float* a_param = (const float*)d_in[12];
  const float* w_out   = (const float*)d_in[13];
  const float* b_out   = (const float*)d_in[14];
  float* out = (float*)d_out;

  const size_t MK = (size_t)M_ * WIDTH_;   // 8192*2560 elements

  size_t off = 0;
  char* wsb = (char*)d_ws;
  auto alloc = [&](size_t bytes) -> char* {
    char* p = wsb + off;
    off += (bytes + 255) & ~(size_t)255;
    return p;
  };
  __bf16* xb    = (__bf16*)alloc(MK * 2);                       // x in bf16
  __bf16* wyt   = (__bf16*)alloc((size_t)WIDTH_ * LRU_ * 2);    // w_y^T bf16
  __bf16* wxt   = (__bf16*)alloc((size_t)WIDTH_ * LRU_ * 2);    // w_x^T bf16
  __bf16* wot   = (__bf16*)alloc((size_t)LRU_ * WIDTH_ * 2);    // w_out^T bf16
  __bf16* wit   = (__bf16*)alloc((size_t)NB_ * BD_ * BD_ * 2);  // wi_gate blk-T
  __bf16* wat   = (__bf16*)alloc((size_t)NB_ * BD_ * BD_ * 2);  // wa_gate blk-T
  float*  yf    = (float*) alloc(MK * 4);                       // gelu branch
  float*  hpre  = (float*) alloc(MK * 4);                       // pre-conv; reused as Gx
  float*  hconv = (float*) alloc(MK * 4);
  __bf16* hcb   = (__bf16*)alloc(MK * 2);                       // hconv bf16
  float*  ga    = (float*) alloc(MK * 4);
  __bf16* hyb   = (__bf16*)alloc(MK * 2);                       // (h*y) bf16
  (void)ws_size; (void)in_sizes; (void)n_in; (void)out_size;

  // --- conversions -------------------------------------------------------
  k_cvt_bf16<<<(int)((MK + 255) / 256), 256, 0, stream>>>(x, xb, (int)MK);
  k_cvt_t<<<(WIDTH_ * LRU_ + 255) / 256, 256, 0, stream>>>(w_y, wyt, WIDTH_, LRU_);
  k_cvt_t<<<(WIDTH_ * LRU_ + 255) / 256, 256, 0, stream>>>(w_x, wxt, WIDTH_, LRU_);
  k_cvt_t<<<(LRU_ * WIDTH_ + 255) / 256, 256, 0, stream>>>(w_out, wot, LRU_, WIDTH_);
  k_cvt_blk_t<<<(NB_ * BD_ * BD_ + 255) / 256, 256, 0, stream>>>(wi_gate, wit);
  k_cvt_blk_t<<<(NB_ * BD_ * BD_ + 255) / 256, 256, 0, stream>>>(wa_gate, wat);

  // fused dual: y = gelu(x@w_y+b_y), hpre = x@w_x+b_x  (LDS-staged, A once)
  k_gemm_lds<1, 1><<<dim3(M_ / 64, LRU_ / 64, 1), 128, 0, stream>>>(
      xb, WIDTH_, wyt, wxt, WIDTH_, b_y, b_x, yf, hpre, LRU_,
      WIDTH_, 0, 0, 0, 0);

  // temporal conv
  k_conv<<<(int)((MK + 255) / 256), 256, 0, stream>>>(hpre, conv_w, conv_b, segp, hconv, hcb);

  // block-diagonal gate GEMMs + sigmoid, batched over the 10 blocks via grid.z
  // (Gx reuses the hpre buffer)
  float* gx = hpre;
  k_gemm_lds<1, 2><<<dim3(M_ / 64, BD_ / 64, NB_), 128, 0, stream>>>(
      hcb, LRU_, wit, wat, BD_, bi_gate, ba_gate, gx, ga, LRU_,
      BD_, BD_, BD_ * BD_, BD_, BD_);

  // RG-LRU scan fused with h*y and bf16 convert
  k_scan<<<(B_ * LRU_ + 255) / 256, 256, 0, stream>>>(gx, ga, hconv, yf, a_param, segp, hyb);

  // out = (h*y) @ w_out + b_out   (LDS-staged)
  k_gemm_lds<0, 0><<<dim3(M_ / 64, WIDTH_ / 64, 1), 128, 0, stream>>>(
      hyb, LRU_, wot, (const __bf16*)0, LRU_, b_out, (const float*)0,
      out, (float*)0, WIDTH_, LRU_, 0, 0, 0, 0);
}
